// PixelMultiClassInfoNCELoss_87479893885192
// MI455X (gfx1250) — matrix operand order repaired and checked by reading, not compiled, not run on hardware
//
#include <hip/hip_runtime.h>
#include <hip/hip_bf16.h>

typedef __attribute__((ext_vector_type(16))) __bf16 v16bf;
typedef __attribute__((ext_vector_type(8)))  __bf16 v8bf;
typedef __attribute__((ext_vector_type(8)))  float  v8f;

#define NPIX 8192
#define CDIM 128
#define PANEL 128           // columns staged per LDS iteration
#define LDS_STRIDE 136      // 128 bf16 + 8 bf16 pad -> 272B row stride (bank-conflict free)
#define EXP_SCALE 14.42695040888963f   // (1/T) * log2(e) = 10 * 1.4426950408889634
#define LN2_F 0.6931471805599453f
#define EPS_F 1e-6f

union Frag { v16bf v; v8bf h[2]; };

// ---------------- Kernel 1: L2-normalize pixel vectors, fp32 -> bf16 ----------------
// emb is (B=2, C=128, H=64, W=64); pixel n = b*4096 + (h*64+w); feature stride = 4096.
__global__ void nrm_kernel(const float* __restrict__ emb, __bf16* __restrict__ En) {
  int n = blockIdx.x * blockDim.x + threadIdx.x;
  if (n >= NPIX) return;
  int b = n >> 12;
  int p = n & 4095;
  const float* base = emb + (size_t)b * (CDIM * 4096) + p;
  float ss = 0.f;
#pragma unroll 8
  for (int c = 0; c < CDIM; ++c) {
    float x = base[(size_t)c << 12];
    ss += x * x;
  }
  float nrm = sqrtf(ss);
  nrm = fmaxf(nrm, 1e-12f);
  float inv = 1.0f / nrm;
  __bf16* dst = En + (size_t)n * CDIM;
#pragma unroll 8
  for (int c = 0; c < CDIM; ++c)
    dst[c] = (__bf16)(base[(size_t)c << 12] * inv);
}

// Per-thread: async-copy 128B (half a panel row) global -> LDS as 8 x 16B chunks.
// Thread's bytes are contiguous in BOTH global and LDS, so offsets advance in lockstep.
__device__ __forceinline__ void stage_panel_async(const __bf16* __restrict__ En,
                                                  __bf16* lds_row_half,
                                                  int p0, int prow, int off) {
  // Low 32 bits of a generic pointer to LDS == LDS byte address (flat aperture truncation).
  unsigned lo = (unsigned)(unsigned long long)(void*)lds_row_half;
  unsigned go = (unsigned)(((unsigned)(p0 + prow) * CDIM + off) * 2);
#pragma unroll
  for (int i = 0; i < 8; ++i) {
    asm volatile("global_load_async_to_lds_b128 %0, %1, %2"
                 :: "v"(lo), "v"(go), "s"(En) : "memory");
    lo += 16;
    go += 16;
  }
}

// ---------------- Kernel 2: fused Gram (bf16 WMMA) + exp + row statistics ----------------
// 8 waves/block; wave w owns rows [ (blockIdx*8+w)*16 , +16 ).
// Column panels double-buffered in LDS, staged with async global->LDS loads.
// Two column tiles are processed with interleaved WMMA chains so consecutive matrix ops
// are independent (no D->C hazard stalls). Diagonal self-term is subtracted at the end,
// recomputed from the resident A fragments (identical bf16 products, fp32 accumulation).
__global__ __launch_bounds__(256, 1)
void gram_kernel(const __bf16* __restrict__ En, const int* __restrict__ lab,
                 float* __restrict__ row_loss) {
  __shared__ __attribute__((aligned(32))) __bf16 panel[2][PANEL * LDS_STRIDE];
  __shared__ int sLab[PANEL];

  const int tid  = threadIdx.x;
  const int wave = tid >> 5;
  const int lane = tid & 31;
  const int half = lane >> 4;          // C layout: half 0 -> rows M=0..7, half 1 -> rows M=8..15
  const int l15  = lane & 15;

  const int r0    = (blockIdx.x * 8 + wave) * 16;   // 16-row tile base
  const int rbase = r0 + half * 8;                  // rows in this lane's accum VGPRs

  const int prow = tid >> 1;           // panel staging: row this thread copies
  const int poff = (tid & 1) * 64;     // and which half of the row (64 bf16 = 128B)

  // Resident A fragments: rows r0..r0+15, K = 0..127.
  // 16-bit A layout (contiguous, matches B): lane<16 holds K {0..15}+k0, lane>=16 holds K {16..31}+k0.
  Frag a[4];
  {
    const __bf16* arow = En + (size_t)(r0 + l15) * CDIM;
    const int kb = half * 16;
#pragma unroll
    for (int kc = 0; kc < 4; ++kc) {
      a[kc].h[0] = *(const v8bf*)(arow + kc * 32 + kb);
      a[kc].h[1] = *(const v8bf*)(arow + kc * 32 + 8 + kb);
    }
  }

  int labR[8];
#pragma unroll
  for (int r = 0; r < 8; ++r) labR[r] = lab[rbase + r];

  float tot[8], pos[8];
#pragma unroll
  for (int r = 0; r < 8; ++r) { tot[r] = 0.f; pos[r] = 0.f; }

  // Prologue: stage panel 0.
  stage_panel_async(En, &panel[0][prow * LDS_STRIDE + poff], 0, prow, poff);

  for (int p0 = 0; p0 < NPIX; p0 += PANEL) {
    const int buf = (p0 / PANEL) & 1;
    __syncthreads();   // everyone done computing on panel[buf^1] -> safe to overwrite

    if (p0 + PANEL < NPIX) {
      stage_panel_async(En, &panel[buf ^ 1][prow * LDS_STRIDE + poff],
                        p0 + PANEL, prow, poff);
      // current panel's 8 async loads complete first (in-order) once <=8 remain
      asm volatile("s_wait_asynccnt 0x8" ::: "memory");
    } else {
      asm volatile("s_wait_asynccnt 0x0" ::: "memory");
    }
    if (tid < PANEL) sLab[tid] = lab[p0 + tid];
    __syncthreads();   // current panel + labels visible to all waves

    const __bf16* pb = &panel[buf][0];
    for (int jt = 0; jt < PANEL / 16; jt += 2) {
      // Two independent column tiles, WMMA chains interleaved.
      const int colL0 = jt * 16 + l15;
      const int colL1 = colL0 + 16;
      // Contiguous 16-bit B layout (matches A): lane = column N, half selects K{0..15} vs K{16..31} per chunk.
      const __bf16* brow0 = pb + colL0 * LDS_STRIDE + half * 16;
      const __bf16* brow1 = pb + colL1 * LDS_STRIDE + half * 16;

      Frag b0[4], b1[4];
#pragma unroll
      for (int kc = 0; kc < 4; ++kc) {
        b0[kc].h[0] = *(const v8bf*)(brow0 + kc * 32);
        b0[kc].h[1] = *(const v8bf*)(brow0 + kc * 32 + 8);
        b1[kc].h[0] = *(const v8bf*)(brow1 + kc * 32);
        b1[kc].h[1] = *(const v8bf*)(brow1 + kc * 32 + 8);
      }

      v8f acc0 = {}, acc1 = {};
#pragma unroll
      for (int kc = 0; kc < 4; ++kc) {
        acc0 = __builtin_amdgcn_wmma_f32_16x16x32_bf16(
            false, a[kc].v, false, b0[kc].v, (short)0, acc0, false, false);
        acc1 = __builtin_amdgcn_wmma_f32_16x16x32_bf16(
            false, a[kc].v, false, b1[kc].v, (short)0, acc1, false, false);
      }

      const int cLab0 = sLab[colL0];
      const int cLab1 = sLab[colL1];
#pragma unroll
      for (int r = 0; r < 8; ++r) {
        float ex0 = __builtin_amdgcn_exp2f(acc0[r] * EXP_SCALE);  // raw v_exp_f32
        float ex1 = __builtin_amdgcn_exp2f(acc1[r] * EXP_SCALE);
        tot[r] += ex0 + ex1;
        pos[r] += ((labR[r] == cLab0) ? ex0 : 0.f) + ((labR[r] == cLab1) ? ex1 : 0.f);
      }
    }
  }

  // Self-similarity of row r0+l15 from this lane's half of the K-range (exact bf16 products).
  float sp = 0.f;
#pragma unroll
  for (int kc = 0; kc < 4; ++kc)
#pragma unroll
    for (int e = 0; e < 16; ++e) {
      float v = (float)a[kc].v[e];
      sp = fmaf(v, v, sp);
    }
  sp += __shfl_xor(sp, 16, 32);        // combine K-halves -> full self-dot of row r0+l15

  // Reduce the 16 lanes that share each row (xor 1,2,4,8 stays within each 16-lane half).
#pragma unroll
  for (int m = 1; m <= 8; m <<= 1) {
#pragma unroll
    for (int r = 0; r < 8; ++r) {
      tot[r] += __shfl_xor(tot[r], m, 32);
      pos[r] += __shfl_xor(pos[r], m, 32);
    }
  }

  // Fetch self-dot of the row this lane will write (all lanes participate in the shuffle).
  float selfdot = __shfl(sp, half * 8 + l15, 32);
  float selfex  = __builtin_amdgcn_exp2f(EXP_SCALE * selfdot);

  if (l15 < 8) {
    const int r = l15;
    float t = tot[r] - selfex;         // remove self-term (reference masks diag to -1e9)
    float p = pos[r] - selfex;         // self label always matches its own row
    row_loss[rbase + r] = -LN2_F * __builtin_amdgcn_logf(p / (t + EPS_F));
  }
}

// ---------------- Kernel 3: per-class means, averaged over present classes ----------------
__global__ void loss_kernel(const float* __restrict__ row_loss, const int* __restrict__ lab,
                            float* __restrict__ out) {
  __shared__ float csum[4];
  __shared__ float ccnt[4];
  if (threadIdx.x < 4) { csum[threadIdx.x] = 0.f; ccnt[threadIdx.x] = 0.f; }
  __syncthreads();
  float ps[4] = {0.f, 0.f, 0.f, 0.f};
  float pc[4] = {0.f, 0.f, 0.f, 0.f};
  for (int n = threadIdx.x; n < NPIX; n += blockDim.x) {
    int l = lab[n];
    float v = row_loss[n];
#pragma unroll
    for (int k = 0; k < 4; ++k) {
      ps[k] += (l == k) ? v : 0.f;
      pc[k] += (l == k) ? 1.f : 0.f;
    }
  }
#pragma unroll
  for (int k = 0; k < 4; ++k) {
    atomicAdd(&csum[k], ps[k]);
    atomicAdd(&ccnt[k], pc[k]);
  }
  __syncthreads();
  if (threadIdx.x == 0) {
    float acc = 0.f, np = 0.f;
    for (int k = 0; k < 4; ++k)
      if (ccnt[k] > 0.f) { acc += csum[k] / ccnt[k]; np += 1.f; }
    out[0] = acc / np;
  }
}

extern "C" void kernel_launch(void* const* d_in, const int* in_sizes, int n_in,
                              void* d_out, int out_size, void* d_ws, size_t ws_size,
                              hipStream_t stream) {
  const float* emb = (const float*)d_in[0];   // (2,128,64,64) fp32
  const int*   lab = (const int*)d_in[1];     // (2,64,64) int32, flat N=8192

  __bf16* En       = (__bf16*)d_ws;                                        // 2 MB
  float*  row_loss = (float*)((char*)d_ws + (size_t)NPIX * CDIM * sizeof(__bf16)); // 32 KB
  float*  out      = (float*)d_out;

  nrm_kernel<<<NPIX / 256, 256, 0, stream>>>(emb, En);
  gram_kernel<<<NPIX / (8 * 16), 256, 0, stream>>>(En, lab, row_loss);
  loss_kernel<<<1, 256, 0, stream>>>(row_loss, lab, out);
}
